// CrossAttention_403726926426
// MI455X (gfx1250) — matrix operand order repaired and checked by reading, not compiled, not run on hardware
//
#include <hip/hip_runtime.h>
#include <hip/hip_bf16.h>

// ---------------------------------------------------------------------------
// Problem constants (from the reference)
// ---------------------------------------------------------------------------
#define BB    8
#define NIMG  4096
#define NTXT  512
#define IMGD  512
#define TXTD  768
#define NH    8
#define HDIM  64

typedef _Float16 half4v  __attribute__((ext_vector_type(4)));
typedef _Float16 half8v  __attribute__((ext_vector_type(8)));
typedef _Float16 half16v __attribute__((ext_vector_type(16)));
typedef float    float8v __attribute__((ext_vector_type(8)));
typedef unsigned int uint4v __attribute__((ext_vector_type(4)));
typedef int      int4v  __attribute__((ext_vector_type(4)));
typedef int      int8v  __attribute__((ext_vector_type(8)));

// Attention LDS budget: K tile 64KB + V tile 64KB + 8 private P tiles (10KB)
#define KV_ELEMS   (NTXT * HDIM)                 // 32768 halfs = 64 KB
#define P_STRIDE   40
#define ATTN_LDS_BYTES ((2 * KV_ELEMS + 8 * 16 * P_STRIDE) * (int)sizeof(_Float16))

// ---------------------------------------------------------------------------
// WMMA helpers (gfx1250: V_WMMA_F32_16X16X32_F16, wave32)
// ---------------------------------------------------------------------------
__device__ __forceinline__ float8v wmma_f16(half16v a, half16v b, float8v c) {
    // (neg_a, A, neg_b, B, c_mod, C, reuse_a, reuse_b)
    return __builtin_amdgcn_wmma_f32_16x16x32_f16(false, a, false, b,
                                                  (short)0, c, false, false);
}

// A-matrix 16x32 fragment (row-major source, ld elements per row).
// Lane l holds row m = l&15; K = (l>>4)*16 + {0..15}.
__device__ __forceinline__ half16v load_a_frag(const _Float16* __restrict__ base,
                                               int ld, int lane) {
    const int row  = lane & 15;
    const int koff = (lane >> 4) * 16;
    half8v lo = *(const half8v*)(base + (size_t)row * ld + koff);
    half8v hi = *(const half8v*)(base + (size_t)row * ld + 8 + koff);
    return __builtin_shufflevector(lo, hi, 0, 1, 2, 3, 4, 5, 6, 7,
                                           8, 9, 10, 11, 12, 13, 14, 15);
}

// B-matrix 32x16 fragment. Source is stored "N rows x K contiguous" (i.e. we
// compute X @ W^T with W row-major [N][K]).  Lane l holds col n = l&15,
// K = (l>>4)*16 + {0..15}, all contiguous -> one 32-byte load.
__device__ __forceinline__ half16v load_b_frag(const _Float16* __restrict__ base,
                                               int ld, int lane) {
    const int col  = lane & 15;
    const int koff = (lane >> 4) * 16;
    return *(const half16v*)(base + (size_t)col * ld + koff);
}

// ---------------------------------------------------------------------------
// Tensor Data Mover: 1-D contiguous DMA global -> LDS.
// D# packed per CDNA5 ISA ch.8: group0 = {count/type/lds_addr/global_addr},
// group1 = {data_size, tensor_dim0/1, tile_dim0..2, strides}.
// units8 = transfer size in 8-byte elements (data_size code 3).
// This toolchain exposes the 6-arg builtin:
//   (uint32x4 g0, int32x8 g1, int32x4, int32x4, int32x8, i32 cpol)
// ---------------------------------------------------------------------------
#if __has_builtin(__builtin_amdgcn_tensor_load_to_lds)
#define HAVE_TDM 1
__device__ __forceinline__ void tdm_load_1d(unsigned lds_off, const void* gptr,
                                            unsigned units8) {
    const unsigned long long ga = (unsigned long long)(uintptr_t)gptr;
    uint4v g0;
    g0.x = 1u;                                    // count=1, user descriptor
    g0.y = lds_off;                               // lds_addr (bytes)
    g0.z = (unsigned)ga;                          // global_addr[31:0]
    g0.w = (unsigned)((ga >> 32) & 0x01FFFFFFu)   // global_addr[56:32]
           | (2u << 30);                          // type = 2 ("image")
    int8v g1;
    g1[0] = (int)(3u << 16);                      // wg_mask=0, data_size=8B
    g1[1] = (int)((units8 & 0xFFFFu) << 16);      // tensor_dim0[15:0]
    g1[2] = (int)(((units8 >> 16) & 0xFFFFu)      // tensor_dim0[31:16]
                  | (1u << 16));                  // tensor_dim1 = 1
    g1[3] = (int)((units8 & 0xFFFFu) << 16);      // tile_dim0 (1-D tile)
    g1[4] = 0;                                    // tile_dim1/2 unused
    g1[5] = (int)units8;                          // tensor_dim0_stride[31:0]
    g1[6] = 0;
    g1[7] = 0;
    int4v z4 = {};                                // groups 2/3 unused (<=2-D)
    int8v z8 = {};
    __builtin_amdgcn_tensor_load_to_lds(g0, g1, z4, z4, z8, 0);
}
#else
#define HAVE_TDM 0
#endif

// ---------------------------------------------------------------------------
// f32 -> f16 convert, 4 elements / lane
// ---------------------------------------------------------------------------
__global__ __launch_bounds__(256) void cvt_f32_f16_x4(
    const float* __restrict__ src, _Float16* __restrict__ dst, int n4) {
    int i = blockIdx.x * 256 + threadIdx.x;
    if (i < n4) {
        float4 v = ((const float4*)src)[i];
        half4v h;
        h.x = (_Float16)v.x; h.y = (_Float16)v.y;
        h.z = (_Float16)v.z; h.w = (_Float16)v.w;
        ((half4v*)dst)[i] = h;
    }
}

// ---------------------------------------------------------------------------
// Generic TN GEMM:  C[M,N] = A[M,K] @ W[N,K]^T  using WMMA f16->f32.
// Block: 256 threads = 8 waves, tile 128(M) x 64(N), 32x32 per wave.
// mode 0: out16 as [b][h][row][d]   (Q / K projections, head split)
// mode 1: out16 as [b][h][d][row]   (V projection, transposed for P·V)
// mode 2: out32 row-major + bias    (final O projection -> d_out)
// ---------------------------------------------------------------------------
__global__ __launch_bounds__(256) void gemm_tn_wmma(
    const _Float16* __restrict__ A, const _Float16* __restrict__ W,
    const float* __restrict__ bias,
    _Float16* __restrict__ out16, float* __restrict__ out32,
    int M, int N, int Kd, int mode, int rowsPerBatch) {
    const int lane = threadIdx.x & 31;
    const int wave = threadIdx.x >> 5;
    const int m0 = blockIdx.x * 128 + (wave >> 1) * 32;
    const int n0 = blockIdx.y * 64  + (wave & 1) * 32;
    if (m0 >= M) return;

    float8v acc00 = {}, acc01 = {}, acc10 = {}, acc11 = {};

    for (int k = 0; k < Kd; k += 32) {
        half16v a0 = load_a_frag(A + (size_t)m0 * Kd + k, Kd, lane);
        half16v a1 = load_a_frag(A + (size_t)(m0 + 16) * Kd + k, Kd, lane);
        half16v b0 = load_b_frag(W + (size_t)n0 * Kd + k, Kd, lane);
        half16v b1 = load_b_frag(W + (size_t)(n0 + 16) * Kd + k, Kd, lane);
        acc00 = wmma_f16(a0, b0, acc00);
        acc01 = wmma_f16(a0, b1, acc01);
        acc10 = wmma_f16(a1, b0, acc10);
        acc11 = wmma_f16(a1, b1, acc11);
    }

    // C/D layout: VGPR r, lanes 0-15 -> M=r, N=lane; lanes 16-31 -> M=8+r.
    const int colA = lane & 15;
    const int rHi  = (lane >> 4) * 8;
    float8v accs[2][2] = {{acc00, acc01}, {acc10, acc11}};
#pragma unroll
    for (int mi = 0; mi < 2; ++mi) {
#pragma unroll
        for (int ni = 0; ni < 2; ++ni) {
#pragma unroll
            for (int r = 0; r < 8; ++r) {
                const int m = m0 + mi * 16 + rHi + r;
                const int n = n0 + ni * 16 + colA;
                const float v = accs[mi][ni][r];
                if (mode == 2) {
                    out32[(size_t)m * N + n] = v + bias[n];
                } else {
                    const int b = m / rowsPerBatch, row = m % rowsPerBatch;
                    const int h = n >> 6, d = n & 63;
                    size_t idx = (mode == 0)
                        ? (((size_t)(b * NH + h) * rowsPerBatch + row) * HDIM + d)
                        : (((size_t)(b * NH + h) * HDIM + d) * rowsPerBatch + row);
                    out16[idx] = (_Float16)v;
                }
            }
        }
    }
}

// ---------------------------------------------------------------------------
// Fused flash-attention: softmax(scale * Q K^T + mask) V
// grid = (B*H, NIMG/128), 256 threads = 8 waves, 16 query rows / wave.
// q  : [B*H][NIMG][64]   k : [B*H][NTXT][64]   vT : [B*H][64][NTXT]
// att: [B][NIMG][H*64]   (f16, feeds the final O projection)
//
// K and V for this head (64KB each) are staged once per block into LDS via
// the Tensor Data Mover (wave 0 issues two tensor_load_to_lds DMAs, waits on
// TENSORcnt, then the workgroup barrier releases the other waves).  All
// B-operand fragments are then served from LDS (ds_load_b128), removing the
// 8x per-wave redundant global fragment traffic.
// ---------------------------------------------------------------------------
__global__ __launch_bounds__(256) void attn_flash_wmma(
    const _Float16* __restrict__ q, const _Float16* __restrict__ k,
    const _Float16* __restrict__ vT, const int* __restrict__ tmask,
    _Float16* __restrict__ att, float scale) {
    extern __shared__ _Float16 smem[];
    _Float16* kLds = smem;                       // [NTXT][64]
    _Float16* vLds = smem + KV_ELEMS;            // [64][NTXT]
    _Float16* pAll = smem + 2 * KV_ELEMS;        // 8 x 16x32 P tiles

    const int lane = threadIdx.x & 31;
    const int wave = threadIdx.x >> 5;
    const int bh = blockIdx.x;
    const int b = bh >> 3, h = bh & 7;
    const int i0 = blockIdx.y * 128 + wave * 16;

    const _Float16* qbase = q + ((size_t)bh * NIMG + i0) * HDIM;
    const _Float16* kbase = k + (size_t)bh * NTXT * HDIM;
    const _Float16* vbase = vT + (size_t)bh * HDIM * NTXT;
    _Float16* plds = pAll + wave * (16 * P_STRIDE);

    // ---- stage K and V head tiles into LDS --------------------------------
#if HAVE_TDM
    if (wave == 0) {
        tdm_load_1d((unsigned)(uintptr_t)kLds, kbase, KV_ELEMS / 4);  // 8B units
        tdm_load_1d((unsigned)(uintptr_t)vLds, vbase, KV_ELEMS / 4);
        __builtin_amdgcn_s_wait_tensorcnt(0);
    }
#else
    for (int idx = threadIdx.x; idx < KV_ELEMS / 8; idx += 256) {
        ((half8v*)kLds)[idx] = ((const half8v*)kbase)[idx];
        ((half8v*)vLds)[idx] = ((const half8v*)vbase)[idx];
    }
#endif
    __syncthreads();

    // Q fragment is register-resident for the whole key sweep (16x64 = 2 frags)
    const half16v qa0 = load_a_frag(qbase, HDIM, lane);
    const half16v qa1 = load_a_frag(qbase + 32, HDIM, lane);

    float8v accO[4] = {};                 // 16 rows x 64 dims
    float m_run[8], l_run[8];
#pragma unroll
    for (int r = 0; r < 8; ++r) { m_run[r] = -1e30f; l_run[r] = 0.0f; }

    const int colA = lane & 15;
    const int rHi  = (lane >> 4) * 8;

    for (int jt = 0; jt < NTXT; jt += 32) {
        // ---- scores: two 16x16 frags over this 32-key chunk -------------
        float8v s[2];
#pragma unroll
        for (int t = 0; t < 2; ++t) {
            const _Float16* kb = kLds + (size_t)(jt + t * 16) * HDIM;
            half16v k0 = load_b_frag(kb, HDIM, lane);       // d = 0..31
            half16v k1 = load_b_frag(kb + 32, HDIM, lane);  // d = 32..63
            float8v z = {};
            z = wmma_f16(qa0, k0, z);
            s[t] = wmma_f16(qa1, k1, z);
            const int j = jt + t * 16 + colA;               // this lane's key
            const float madd = tmask[b * NTXT + j] ? 0.0f : -1e30f;
#pragma unroll
            for (int r = 0; r < 8; ++r) s[t][r] = s[t][r] * scale + madd;
        }

        // ---- online softmax (rows live across 16 lanes of a half-wave) ---
        float mnew[8], alpha[8];
#pragma unroll
        for (int r = 0; r < 8; ++r) {
            float t = fmaxf(s[0][r], s[1][r]);
#pragma unroll
            for (int mk = 8; mk >= 1; mk >>= 1)
                t = fmaxf(t, __shfl_xor(t, mk, 16));
            mnew[r]  = fmaxf(m_run[r], t);
            alpha[r] = __expf(m_run[r] - mnew[r]);
            m_run[r] = mnew[r];
        }
#pragma unroll
        for (int r = 0; r < 8; ++r) {
            float p0 = __expf(s[0][r] - mnew[r]);
            float p1 = __expf(s[1][r] - mnew[r]);
            s[0][r] = p0; s[1][r] = p1;
            float ts = p0 + p1;
#pragma unroll
            for (int mk = 8; mk >= 1; mk >>= 1)
                ts += __shfl_xor(ts, mk, 16);
            l_run[r] = l_run[r] * alpha[r] + ts;
        }
#pragma unroll
        for (int dt = 0; dt < 4; ++dt)
#pragma unroll
            for (int r = 0; r < 8; ++r) accO[dt][r] *= alpha[r];

        // ---- re-shape P: C-layout -> A-layout via private LDS slice ------
#pragma unroll
        for (int t = 0; t < 2; ++t)
#pragma unroll
            for (int r = 0; r < 8; ++r)
                plds[(rHi + r) * P_STRIDE + t * 16 + colA] = (_Float16)s[t][r];
        // same-wave LDS RAW is hardware-ordered (DScnt in-order); the fence
        // only stops the compiler from reordering around it.
        asm volatile("s_wait_dscnt 0" ::: "memory");
        half16v pa = load_a_frag(plds, P_STRIDE, lane);

        // ---- accumulate P(16x32) . V(32x64): 4 WMMAs --------------------
#pragma unroll
        for (int dt = 0; dt < 4; ++dt) {
            half16v vb = load_b_frag(vLds + (size_t)dt * 16 * NTXT + jt,
                                     NTXT, lane);
            accO[dt] = wmma_f16(pa, vb, accO[dt]);
        }
    }

    // ---- normalize and store attended (f16, row-major [B][NIMG][512]) ----
    float rinv[8];
#pragma unroll
    for (int r = 0; r < 8; ++r)
        rinv[r] = (l_run[r] > 0.0f) ? (1.0f / l_run[r]) : 0.0f;
#pragma unroll
    for (int dt = 0; dt < 4; ++dt)
#pragma unroll
        for (int r = 0; r < 8; ++r) {
            const int i = i0 + rHi + r;
            const int d = dt * 16 + colA;
            att[((size_t)b * NIMG + i) * IMGD + h * HDIM + d] =
                (_Float16)(accO[dt][r] * rinv[r]);
        }
}

// ---------------------------------------------------------------------------
// Host-side orchestration
// ---------------------------------------------------------------------------
extern "C" void kernel_launch(void* const* d_in, const int* in_sizes, int n_in,
                              void* d_out, int out_size, void* d_ws, size_t ws_size,
                              hipStream_t stream) {
    const float* img   = (const float*)d_in[0];
    const float* text  = (const float*)d_in[1];
    const int*   tmask = (const int*)d_in[2];
    const float* Wq    = (const float*)d_in[3];
    const float* Wk    = (const float*)d_in[4];
    const float* Wv    = (const float*)d_in[5];
    const float* Wo    = (const float*)d_in[6];
    const float* bo    = (const float*)d_in[7];
    float* out = (float*)d_out;

    // workspace carve (all f16); att16 aliases img16 (dead after Q-proj)
    char* p = (char*)d_ws;
    auto carve = [&](size_t nelem) {
        _Float16* r = (_Float16*)p;
        p += nelem * sizeof(_Float16);
        return r;
    };
    _Float16* img16  = carve((size_t)BB * NIMG * IMGD);   // 33.5 MB
    _Float16* text16 = carve((size_t)BB * NTXT * TXTD);   //  6.3 MB
    _Float16* Wq16   = carve((size_t)IMGD * IMGD);
    _Float16* Wk16   = carve((size_t)IMGD * TXTD);
    _Float16* Wv16   = carve((size_t)IMGD * TXTD);
    _Float16* Wo16   = carve((size_t)IMGD * IMGD);
    _Float16* q16    = carve((size_t)BB * NH * NIMG * HDIM);
    _Float16* k16    = carve((size_t)BB * NH * NTXT * HDIM);
    _Float16* vT16   = carve((size_t)BB * NH * HDIM * NTXT);
    _Float16* att16  = img16;                              // alias

    auto cvt = [&](const float* s, _Float16* d, size_t n) {
        int n4 = (int)(n / 4);
        cvt_f32_f16_x4<<<(n4 + 255) / 256, 256, 0, stream>>>(s, d, n4);
    };
    cvt(img,  img16,  (size_t)BB * NIMG * IMGD);
    cvt(text, text16, (size_t)BB * NTXT * TXTD);
    cvt(Wq, Wq16, (size_t)IMGD * IMGD);
    cvt(Wk, Wk16, (size_t)IMGD * TXTD);
    cvt(Wv, Wv16, (size_t)IMGD * TXTD);
    cvt(Wo, Wo16, (size_t)IMGD * IMGD);

    // Q = img @ Wq^T   -> [b][h][i][d]
    {
        dim3 g((BB * NIMG + 127) / 128, IMGD / 64);
        gemm_tn_wmma<<<g, 256, 0, stream>>>(img16, Wq16, nullptr, q16, nullptr,
                                            BB * NIMG, IMGD, IMGD, 0, NIMG);
    }
    // K = text @ Wk^T  -> [b][h][j][d]
    {
        dim3 g((BB * NTXT + 127) / 128, IMGD / 64);
        gemm_tn_wmma<<<g, 256, 0, stream>>>(text16, Wk16, nullptr, k16, nullptr,
                                            BB * NTXT, IMGD, TXTD, 0, NTXT);
    }
    // V = text @ Wv^T  -> transposed [b][h][d][j]
    {
        dim3 g((BB * NTXT + 127) / 128, IMGD / 64);
        gemm_tn_wmma<<<g, 256, 0, stream>>>(text16, Wv16, nullptr, vT16, nullptr,
                                            BB * NTXT, IMGD, TXTD, 1, NTXT);
    }
    // fused attention -> att16 (aliases img16)
    {
        const float scale = 0.10206207261596577f;   // (768/8)^-0.5
        dim3 g(BB * NH, NIMG / 128);
        attn_flash_wmma<<<g, 256, ATTN_LDS_BYTES, stream>>>(q16, k16, vT16,
                                                            tmask, att16, scale);
    }
    // out = att @ Wo^T + bo   (f32)
    {
        dim3 g((BB * NIMG + 127) / 128, IMGD / 64);
        gemm_tn_wmma<<<g, 256, 0, stream>>>(att16, Wo16, bo, nullptr, out,
                                            BB * NIMG, IMGD, IMGD, 2, 0);
    }
}
